// TLC_GraphAgent_48533130445277
// MI455X (gfx1250) — matrix operand order repaired and checked by reading, not compiled, not run on hardware
//
#include <hip/hip_runtime.h>
#include <hip/hip_bf16.h>

// ---------------------------------------------------------------------------
// Shapes from the reference. Tile math assumes N % 64 == 0 (N = 200000 ok).
// ---------------------------------------------------------------------------
#define DIN   275
#define DINP  288          // DIN padded to multiple of 32 for WMMA K loop
#define H     64
#define H3    192          // 3*H (GRU gates)
#define NA    14           // actions

typedef __bf16 v16bf __attribute__((ext_vector_type(16)));
typedef float  v8f   __attribute__((ext_vector_type(8)));

// ---------------------------------------------------------------------------
// WMMA register layouts (CDNA5 ISA §7.12.2), wave32:
//   A 16x32 bf16 : row = lane&15, k = e + (e&8) + 8*(lane>=16)   e=0..15
//   B 32x16 bf16 : col = lane&15, k = e + (e&8) + 8*(lane>=16)
//   C/D 16x16 f32: row = v + 8*(lane>=16), col = lane&15
//
// bf16 activation buffers are stored A-register-native ("swizzled"):
//   element (row tile t, r = row%16, feature k):
//     kc = k>>5, kk = k&31, hi = (kk>>3)&1, e = (kk&7) + ((kk>>4)<<3)
//     index = t*1024 + kc*512 + (hi*16 + r)*16 + e
// so a wave loads its whole A k-chunk as one 32-byte (2 x b128) vector.
// ---------------------------------------------------------------------------
__device__ __forceinline__ long a_swz(long tile, int r, int k) {
    int kc = k >> 5;
    int kk = k & 31;
    int hi = (kk >> 3) & 1;
    int e  = (kk & 7) + ((kk >> 4) << 3);
    return tile * 1024 + kc * 512 + (hi * 16 + r) * 16 + e;
}

// ---------------------------------------------------------------------------
// Weight pack: W [Mw, Kw] f32 row-major -> P in B-register-native order:
//   P[((kc*MT + c)*32 + lane)*16 + e] = W^T[kc*32 + e + (e&8) + 8*hi][c*16 + mn]
// Zero padded for k >= Kw.
// ---------------------------------------------------------------------------
__global__ void k_pack_wB(const float* __restrict__ W, __bf16* __restrict__ P,
                          int Mw, int Kw, int Kpad) {
    int total = (Kpad >> 5) * (Mw >> 4) * 512;
    int g = blockIdx.x * blockDim.x + threadIdx.x;
    if (g >= total) return;
    int e    = g & 15;
    int lane = (g >> 4) & 31;
    int ct   = g >> 9;                 // kc*MT + c
    int MT   = Mw >> 4;
    int kc   = ct / MT;
    int c    = ct - kc * MT;
    int hi   = lane >> 4;
    int mn   = lane & 15;
    int k    = kc * 32 + e + (e & 8) + hi * 8;
    int m    = c * 16 + mn;
    P[g] = (k < Kw) ? (__bf16)W[(long)m * Kw + k] : (__bf16)0.0f;
}

// ---------------------------------------------------------------------------
// Degree / normalization
// ---------------------------------------------------------------------------
__global__ void k_init_deg(float* __restrict__ deg, int n) {
    int g = blockIdx.x * blockDim.x + threadIdx.x;
    if (g < n) deg[g] = 1.0f;                       // self loop contributes 1
}

__global__ void k_edge_deg(const long long* __restrict__ ei, float* __restrict__ deg,
                           int E) {
    int e = blockIdx.x * blockDim.x + threadIdx.x;
    if (e >= E) return;
    int dst = (int)ei[(long)E + e];
    atomicAdd(&deg[dst], 1.0f);
}

__global__ void k_dis(const float* __restrict__ deg, float* __restrict__ dis, int n) {
    int g = blockIdx.x * blockDim.x + threadIdx.x;
    if (g < n) dis[g] = rsqrtf(deg[g]);             // deg >= 1 always (self loop)
}

// ---------------------------------------------------------------------------
// WMMA GEMM:  out[N, M] = A[N, KREAL] * W^T  (+bias) (relu)
//   A_F32:  A is f32 row-major, converted to bf16 in-register (inputs/hidden).
//   !A_F32: A is a bf16 activation buffer in A-register-native swizzle
//           -> one v16bf (2 x b128) load per k-chunk.
//   Bp is bf16 in B-register-native order -> one v16bf load per tile.
//   One wave computes a 16-row x M stripe (MT accumulators); A is reused
//   across all MT column tiles. nrows % 64 == 0 so EXEC stays all-ones.
// ---------------------------------------------------------------------------
template <int M, int KREAL, int KPAD, bool A_F32, bool BIAS, bool RELU,
          bool OUT_F32, bool OUT_BF16>
__global__ __launch_bounds__(128) void k_gemm(
    const void* __restrict__ Av, const __bf16* __restrict__ Bp,
    const float* __restrict__ bias, float* __restrict__ outF,
    __bf16* __restrict__ outB) {
    constexpr int MT  = M / 16;
    constexpr int NKC = KPAD / 32;
    const int lane = threadIdx.x & 31;
    const int wave = threadIdx.x >> 5;
    const int hi   = lane >> 4;
    const int mn   = lane & 15;
    const long tile    = (long)blockIdx.x * 4 + wave;
    const long rowBase = tile * 16;

    const float*  Af = (const float*)Av;
    const __bf16* Ab = (const __bf16*)Av;

    v8f acc[MT];
#pragma unroll
    for (int c = 0; c < MT; ++c) acc[c] = (v8f){0, 0, 0, 0, 0, 0, 0, 0};

#pragma unroll
    for (int kc = 0; kc < NKC; ++kc) {
        // ---- A tile ----
        v16bf a;
        if (A_F32) {
#pragma unroll
            for (int e = 0; e < 16; ++e) {
                int k = kc * 32 + e + (e & 8) + hi * 8;
                __bf16 v = (__bf16)0.0f;
                if (k < KREAL) v = (__bf16)Af[(rowBase + mn) * (long)KREAL + k];
                a[e] = v;
            }
        } else {
            a = *(const v16bf*)(Ab + tile * 1024 + kc * 512 + lane * 16);
        }
        // ---- MT column tiles of B, one WMMA each ----
#pragma unroll
        for (int c = 0; c < MT; ++c) {
            v16bf b = *(const v16bf*)(Bp + (((long)kc * MT + c) << 9) + lane * 16);
            acc[c] = __builtin_amdgcn_wmma_f32_16x16x32_bf16(
                false, a, false, b, (short)0, acc[c], false, false);
        }
    }

    // ---- epilogue: bias / relu / store (f32 row-major and/or bf16 swizzled) ----
#pragma unroll
    for (int c = 0; c < MT; ++c) {
#pragma unroll
        for (int v = 0; v < 8; ++v) {
            int r    = v + 8 * hi;                // row within tile
            int ocol = c * 16 + mn;
            float x = acc[c][v];
            if (BIAS) x += bias[ocol];
            if (RELU) x = fmaxf(x, 0.0f);
            if (OUT_F32)  outF[(rowBase + r) * (long)M + ocol] = x;
            if (OUT_BF16) outB[a_swz(tile, r, ocol)] = (__bf16)x;
        }
    }
}

// ---------------------------------------------------------------------------
// GRU gates: h2 = (1-z)*n + z*h   (gate order r,z,n; torch GRUCell layout)
// h2 written f32 row-major (model output) and bf16 A-swizzled (next GEMM).
// ---------------------------------------------------------------------------
__global__ void k_gru_gates(const float* __restrict__ gi, const float* __restrict__ gh,
                            const float* __restrict__ hin, float* __restrict__ h2f,
                            __bf16* __restrict__ h2b, int total) {
    int g = blockIdx.x * blockDim.x + threadIdx.x;
    if (g >= total) return;
    int n = g >> 6;
    int f = g & 63;
    long b = (long)n * H3 + f;
    float ir = gi[b], iz = gi[b + H], in_ = gi[b + 2 * H];
    float hr = gh[b], hz = gh[b + H], hn  = gh[b + 2 * H];
    float r = 1.0f / (1.0f + __expf(-(ir + hr)));
    float z = 1.0f / (1.0f + __expf(-(iz + hz)));
    float nn = tanhf(in_ + r * hn);
    float out = (1.0f - z) * nn + z * hin[g];
    h2f[g] = out;
    h2b[a_swz(n >> 4, n & 15, f)] = (__bf16)out;
}

// ---------------------------------------------------------------------------
// Zero an f32 buffer
// ---------------------------------------------------------------------------
__global__ void k_zero(float* __restrict__ p, long n) {
    long g = (long)blockIdx.x * blockDim.x + threadIdx.x;
    if (g < n) p[g] = 0.0f;
}

// ---------------------------------------------------------------------------
// Edge aggregation: agg[dst] += t[src] * dis[src]*dis[dst]
// 64 threads (2 waves) per edge, one feature each. The N x 64 f32 feature
// matrix (51 MB) is L2-resident (192 MB L2), so the random gather and f32
// atomic scatter are mostly L2 traffic, not HBM.
// ---------------------------------------------------------------------------
__global__ void k_edge_agg(const long long* __restrict__ ei,
                           const float* __restrict__ dis,
                           const float* __restrict__ t, float* __restrict__ agg,
                           int E) {
    long g = (long)blockIdx.x * blockDim.x + threadIdx.x;
    int e = (int)(g >> 6);
    int f = (int)(g & 63);
    if (e >= E) return;
    int s = (int)ei[e];
    int d = (int)ei[(long)E + e];
    float w = dis[s] * dis[d];
    atomicAdd(&agg[(long)d * H + f], t[(long)s * H + f] * w);
}

// ---------------------------------------------------------------------------
// GCN finalize: out = agg + dis^2 * t (self loop) + bias, optional relu.
// bf16 output goes to the A-swizzled layout for the next GEMM.
// ---------------------------------------------------------------------------
template <bool RELU, bool OUT_F32, bool OUT_BF16>
__global__ void k_gcn_fin(const float* __restrict__ agg, const float* __restrict__ t,
                          const float* __restrict__ dis, const float* __restrict__ bias,
                          float* __restrict__ outF, __bf16* __restrict__ outB,
                          int total) {
    int g = blockIdx.x * blockDim.x + threadIdx.x;
    if (g >= total) return;
    int n = g >> 6;
    int f = g & 63;
    float di = dis[n];
    float x = agg[g] + di * di * t[g] + bias[f];
    if (RELU) x = fmaxf(x, 0.0f);
    if (OUT_F32)  outF[g] = x;
    if (OUT_BF16) outB[a_swz(n >> 4, n & 15, f)] = (__bf16)x;
}

// ---------------------------------------------------------------------------
// Q head: q[n,a] = h3[n,:] . q_w[a,:] + q_b[a]   (q_w is 14x64, cache-resident)
// ---------------------------------------------------------------------------
__global__ void k_qhead(const float* __restrict__ h3, const float* __restrict__ qw,
                        const float* __restrict__ qb, float* __restrict__ q,
                        int total) {
    int g = blockIdx.x * blockDim.x + threadIdx.x;
    if (g >= total) return;
    int n = g / NA;
    int a = g - n * NA;
    const float* hr = h3 + (long)n * H;
    const float* wr = qw + a * H;
    float s = qb[a];
#pragma unroll
    for (int i = 0; i < H; ++i) s += hr[i] * wr[i];
    q[g] = s;
}

// ---------------------------------------------------------------------------
// Launch
// ---------------------------------------------------------------------------
extern "C" void kernel_launch(void* const* d_in, const int* in_sizes, int n_in,
                              void* d_out, int out_size, void* d_ws, size_t ws_size,
                              hipStream_t stream) {
    const float*     inputs = (const float*)d_in[0];
    const float*     hidden = (const float*)d_in[1];
    const long long* ei     = (const long long*)d_in[2];
    const float*     enc_w  = (const float*)d_in[3];
    const float*     enc_b  = (const float*)d_in[4];
    const float*     w_ih   = (const float*)d_in[5];
    const float*     w_hh   = (const float*)d_in[6];
    const float*     b_ih   = (const float*)d_in[7];
    const float*     b_hh   = (const float*)d_in[8];
    const float*     g1_w   = (const float*)d_in[9];
    const float*     g1_b   = (const float*)d_in[10];
    const float*     g2_w   = (const float*)d_in[11];
    const float*     g2_b   = (const float*)d_in[12];
    const float*     q_w    = (const float*)d_in[13];
    const float*     q_b    = (const float*)d_in[14];

    const int N = in_sizes[1] / H;       // 200000
    const int E = in_sizes[2] / 2;       // 3200000

    float* q_out = (float*)d_out;                       // [N, 14]
    float* h2f   = (float*)d_out + (long)N * NA;        // [N, 64] (2nd output)

    // ---- workspace layout (all offsets 256B-aligned) ----
    char* ws = (char*)d_ws;
    __bf16* encP = (__bf16*)(ws + 0);                     // 9*4*512  bf16
    __bf16* wihP = (__bf16*)(ws + 36864);                 // 2*12*512 bf16
    __bf16* whhP = (__bf16*)(ws + 61440);
    __bf16* g1P  = (__bf16*)(ws + 86016);                 // 2*4*512  bf16
    __bf16* g2P  = (__bf16*)(ws + 94208);
    float*  deg  = (float*) (ws + 102400);                // [N]
    float*  dis  = (float*) (ws + 902400);                // [N]
    __bf16* h1b  = (__bf16*)(ws + 1702400);               // [N,64] bf16 swizzled
    __bf16* h2b  = (__bf16*)(ws + 27302400);              // [N,64] bf16 swizzled
    __bf16* xb   = (__bf16*)(ws + 52902400);              // [N,64] bf16 swizzled
    float*  gi   = (float*) (ws + 78502400);              // [N,192]
    float*  gh   = (float*) (ws + 232102400);             // [N,192]
    // phase-B reuse of the gi/gh regions (gates consumed by then):
    float*  t1   = gi;                                    // [N,64]
    float*  agg1 = (float*)(ws + 78502400 + 51200000);    // [N,64]
    float*  t2   = (float*)(ws + 78502400 + 102400000);   // [N,64]
    float*  agg2 = gh;                                    // [N,64] -> h3 in place

    const int NH   = N * H;
    const int rowB = N / 64;             // gemm blocks: 4 waves x 16 rows

    // ---- weight packing (tiny) ----
    k_pack_wB<<<(9 * 4 * 512 + 255) / 256, 256, 0, stream>>>(enc_w, encP, H, DIN, DINP);
    k_pack_wB<<<(2 * 12 * 512 + 255) / 256, 256, 0, stream>>>(w_ih, wihP, H3, H, H);
    k_pack_wB<<<(2 * 12 * 512 + 255) / 256, 256, 0, stream>>>(w_hh, whhP, H3, H, H);
    k_pack_wB<<<(2 * 4 * 512 + 255) / 256, 256, 0, stream>>>(g1_w, g1P, H, H, H);
    k_pack_wB<<<(2 * 4 * 512 + 255) / 256, 256, 0, stream>>>(g2_w, g2P, H, H, H);

    // ---- degree / normalization ----
    k_init_deg<<<(N + 255) / 256, 256, 0, stream>>>(deg, N);
    k_edge_deg<<<(E + 255) / 256, 256, 0, stream>>>(ei, deg, E);
    k_dis<<<(N + 255) / 256, 256, 0, stream>>>(deg, dis, N);

    // ---- encoder: h1 = relu(inputs @ enc_w.T + enc_b), bf16 swizzled out ----
    k_gemm<H, DIN, DINP, true, true, true, false, true>
        <<<rowB, 128, 0, stream>>>(inputs, encP, enc_b, nullptr, h1b);

    // ---- GRU gate pre-activations ----
    k_gemm<H3, H, H, false, true, false, true, false>
        <<<rowB, 128, 0, stream>>>(h1b, wihP, b_ih, gi, nullptr);
    k_gemm<H3, H, H, true, true, false, true, false>
        <<<rowB, 128, 0, stream>>>(hidden, whhP, b_hh, gh, nullptr);
    k_gru_gates<<<(NH + 255) / 256, 256, 0, stream>>>(gi, gh, hidden, h2f, h2b, NH);

    // ---- GCN layer 1 ----
    k_gemm<H, H, H, false, false, false, true, false>
        <<<rowB, 128, 0, stream>>>(h2b, g1P, nullptr, t1, nullptr);
    k_zero<<<(NH + 255) / 256, 256, 0, stream>>>(agg1, NH);
    k_edge_agg<<<(int)(((long)E * H + 255) / 256), 256, 0, stream>>>(ei, dis, t1, agg1, E);
    k_gcn_fin<true, false, true><<<(NH + 255) / 256, 256, 0, stream>>>(
        agg1, t1, dis, g1_b, nullptr, xb, NH);

    // ---- GCN layer 2 ----
    k_gemm<H, H, H, false, false, false, true, false>
        <<<rowB, 128, 0, stream>>>(xb, g2P, nullptr, t2, nullptr);
    k_zero<<<(NH + 255) / 256, 256, 0, stream>>>(agg2, NH);
    k_edge_agg<<<(int)(((long)E * H + 255) / 256), 256, 0, stream>>>(ei, dis, t2, agg2, E);
    k_gcn_fin<false, true, false><<<(NH + 255) / 256, 256, 0, stream>>>(
        agg2, t2, dis, g2_b, agg2, nullptr, NH);   // h3 written in place

    // ---- Q head ----
    k_qhead<<<(N * NA + 255) / 256, 256, 0, stream>>>(agg2, q_w, q_b, q_out, N * NA);
}
